// Attn_10849087390322
// MI455X (gfx1250) — matrix operand order repaired and checked by reading, not compiled, not run on hardware
//
#include <hip/hip_runtime.h>

// ---------------------------------------------------------------------------
// Types
// ---------------------------------------------------------------------------
typedef _Float16 half_t;
typedef __attribute__((ext_vector_type(16))) _Float16 v16h;
typedef __attribute__((ext_vector_type(8)))  _Float16 v8h;
typedef __attribute__((ext_vector_type(8)))  float    v8f;
typedef unsigned int u32x4 __attribute__((ext_vector_type(4)));
typedef int          i32x8 __attribute__((ext_vector_type(8)));
typedef int          i32x4 __attribute__((ext_vector_type(4)));

#define L_SEQ 4096
#define DMODEL 512
#define NHEAD 8
#define DHEAD 64
// 1/sqrt(DHEAD) * log2(e): scores computed in log2 domain so softmax can use
// the native exp2 hardware op (v_exp_f32) without the ln->log2 multiply.
#define QSCALE 0.18033688011112042f

// ---------------------------------------------------------------------------
// WMMA helpers (gfx1250 wave32 fragment layouts, ISA 7.12.2)
// ---------------------------------------------------------------------------
__device__ __forceinline__ v16h cat8(v8h lo, v8h hi) {
  return __builtin_shufflevector(lo, hi, 0,1,2,3,4,5,6,7,8,9,10,11,12,13,14,15);
}

// A matrix 16x32 f16. base points at (row0, k0) of a row-major [.., ld] array.
// lanes 0-15: M=lane, K={0..7,16..23}; lanes 16-31: M=lane-16, K={8..15,24..31}
__device__ __forceinline__ v16h load_a_frag(const half_t* base, int ld, int lane) {
  const half_t* r = base + (lane & 15) * ld + ((lane & 16) ? 8 : 0);
  v8h lo = *(const v8h*)(r);
  v8h hi = *(const v8h*)(r + 16);
  return cat8(lo, hi);
}

// B matrix 32x16 f16 from an n-major array: element (K,N) at base[N*ld + K].
// lanes 0-15: N=lane, K={0..7,16..23}; lanes 16-31: N=lane-16, K={8..15,24..31}
__device__ __forceinline__ v16h load_b_frag(const half_t* base, int ld, int lane) {
  const half_t* r = base + (lane & 15) * ld + ((lane & 16) ? 8 : 0);
  v8h lo = *(const v8h*)(r);
  v8h hi = *(const v8h*)(r + 16);
  return cat8(lo, hi);
}

__device__ __forceinline__ v8f wmma_f16(v16h a, v16h b, v8f c) {
  return __builtin_amdgcn_wmma_f32_16x16x32_f16(false, a, false, b, (short)0, c,
                                                false, false);
}

// ---------------------------------------------------------------------------
// Tensor Data Mover: 2D tile (f16 elements) global -> LDS. clang-23 6-arg form.
// Descriptor bit packing per CDNA5 ISA.
// ---------------------------------------------------------------------------
__device__ __forceinline__ void tdm_load_2d_f16(unsigned lds_off, const void* gptr,
                                                unsigned tile_w, unsigned tile_h,
                                                unsigned row_stride_elems,
                                                unsigned tensor_w, unsigned tensor_h) {
  unsigned long long ga = (unsigned long long)gptr;
  u32x4 g0 = {
    1u,                                                   // count=1 (valid user D#)
    lds_off,                                              // lds_addr [63:32]
    (unsigned)ga,                                         // global_addr [95:64]
    ((unsigned)((ga >> 32) & 0x01FFFFFFu)) | 0x80000000u  // addr[56:32] | type=2
  };
  i32x8 g1 = {
    (int)0x00010000u,                                           // data_size=1 (2B)
    (int)((tensor_w & 0xFFFFu) << 16),                          // tensor_dim0[15:0]
    (int)(((tensor_w >> 16) & 0xFFFFu) | ((tensor_h & 0xFFFFu) << 16)),
    (int)(((tensor_h >> 16) & 0xFFFFu) | ((tile_w & 0xFFFFu) << 16)),
    (int)(tile_h & 0xFFFFu),                                    // tile_dim1, tile_dim2=0
    (int)row_stride_elems,                                      // tensor_dim0_stride lo
    0, 0
  };
  i32x4 z4 = {0, 0, 0, 0};
  i32x8 z8 = {0, 0, 0, 0, 0, 0, 0, 0};
  __builtin_amdgcn_tensor_load_to_lds(g0, g1, z4, z4, z8, 0);
}

// ---------------------------------------------------------------------------
// Kernel 0a: f32 -> f16 elementwise
// ---------------------------------------------------------------------------
__global__ void cvt_f16_kernel(const float* __restrict__ src,
                               half_t* __restrict__ dst, int n) {
  int i = blockIdx.x * blockDim.x + threadIdx.x;
  if (i < n) dst[i] = (half_t)src[i];
}

// Kernel 0b: f32 [K][N] -> f16 [N][K] (transpose + convert)
__global__ void cvt_tr_kernel(const float* __restrict__ src,
                              half_t* __restrict__ dst, int K, int N) {
  int i = blockIdx.x * blockDim.x + threadIdx.x;
  if (i < K * N) {
    int k = i / N, n = i % N;
    dst[n * K + k] = (half_t)src[i];
  }
}

// ---------------------------------------------------------------------------
// Kernel 1: QKV = xh[4096,512] @ W (Wt is n-major f16 [1536][512]).
// Each wave: 16 rows x 64 cols. Scatter into Qh/Kh (head-major) and V^T.
// Q gets the 1/sqrt(DH)*log2(e) scale folded in (exp2-domain softmax).
// ---------------------------------------------------------------------------
__global__ __launch_bounds__(256) void qkv_gemm_kernel(
    const half_t* __restrict__ xh, const half_t* __restrict__ Wt,
    half_t* __restrict__ Qh, half_t* __restrict__ Kh, half_t* __restrict__ VTh) {
  const int NCH = (3 * DMODEL) / 64;  // 24 column chunks
  int wid = blockIdx.x * 8 + (threadIdx.x >> 5);
  int lane = threadIdx.x & 31;
  int mt = wid / NCH;
  int nc = wid % NCH;
  int row0 = mt * 16;

  v8f acc[4] = {{}, {}, {}, {}};
  for (int k0 = 0; k0 < DMODEL; k0 += 32) {
    v16h a = load_a_frag(xh + row0 * DMODEL + k0, DMODEL, lane);
#pragma unroll
    for (int t = 0; t < 4; ++t) {
      v16h b = load_b_frag(Wt + (nc * 64 + t * 16) * DMODEL + k0, DMODEL, lane);
      acc[t] = wmma_f16(a, b, acc[t]);
    }
  }

  int n = lane & 15;
  int mb = (lane & 16) ? 8 : 0;
  int which = nc >> 3;   // 0=Q 1=K 2=V
  int hh = nc & 7;       // head
#pragma unroll
  for (int t = 0; t < 4; ++t) {
#pragma unroll
    for (int r = 0; r < 8; ++r) {
      int row = row0 + mb + r;
      int d = t * 16 + n;
      float v = acc[t][r];
      if (which == 0) {
        Qh[(hh * L_SEQ + row) * DHEAD + d] = (half_t)(v * QSCALE);
      } else if (which == 1) {
        Kh[(hh * L_SEQ + row) * DHEAD + d] = (half_t)v;
      } else {
        VTh[(hh * DHEAD + d) * L_SEQ + row] = (half_t)v;  // transposed V
      }
    }
  }
}

// ---------------------------------------------------------------------------
// Kernel 2: flash attention. Grid (L/128, H), block 256 (8 waves).
// Each wave owns 16 q rows. 64-row K/V context blocks are staged in LDS via
// the Tensor Data Mover with ping-pong double buffering: block i+1's DMA is
// issued before block i is computed, so the DMA overlaps WMMA+softmax, and
// s_wait_tensorcnt(2) (in-order TDM completion) guarantees block i residency.
// ---------------------------------------------------------------------------
__global__ __launch_bounds__(256) void flash_attn_kernel(
    const half_t* __restrict__ Qh, const half_t* __restrict__ Kh,
    const half_t* __restrict__ VTh, half_t* __restrict__ Oh) {
  __shared__ half_t lds_k[2][64 * DHEAD];    // [buf][kr][d]
  __shared__ half_t lds_v[2][DHEAD * 64];    // [buf][d][kr]  (from V^T)
  __shared__ half_t lds_p[8 * 16 * 32];      // per-wave P staging (C->A layout)

  int h = blockIdx.y;
  int wave = threadIdx.x >> 5;
  int lane = threadIdx.x & 31;
  int qbase = blockIdx.x * 128 + wave * 16;

  const half_t* Q = Qh + (h * L_SEQ + qbase) * DHEAD;
  v16h qa0 = load_a_frag(Q, DHEAD, lane);        // K(d)=0..31
  v16h qa1 = load_a_frag(Q + 32, DHEAD, lane);   // K(d)=32..63

  v8f o[4] = {{}, {}, {}, {}};
  float mr[8], lr[8];
#pragma unroll
  for (int r = 0; r < 8; ++r) { mr[r] = -1e30f; lr[r] = 0.f; }

  half_t* pp = lds_p + wave * (16 * 32);
  int n = lane & 15;
  int mb = (lane & 16) ? 8 : 0;

  unsigned k_off[2] = {(unsigned)(size_t)&lds_k[0][0], (unsigned)(size_t)&lds_k[1][0]};
  unsigned v_off[2] = {(unsigned)(size_t)&lds_v[0][0], (unsigned)(size_t)&lds_v[1][0]};

  const half_t* Kbase = Kh + h * L_SEQ * DHEAD;
  const half_t* Vbase = VTh + h * DHEAD * L_SEQ;

  // prologue: start DMA of block 0
  if (wave == 0) {
    tdm_load_2d_f16(k_off[0], Kbase, DHEAD, 64, DHEAD, DHEAD, 64);
    tdm_load_2d_f16(v_off[0], Vbase, 64, DHEAD, L_SEQ, 64, DHEAD);
  }

  const int NBLK = L_SEQ / 64;
  for (int ib = 0; ib < NBLK; ++ib) {
    int cur = ib & 1;
    if (wave == 0) {
      if (ib + 1 < NBLK) {
        // issue next block into the other buffer, then wait until only those
        // two loads remain outstanding => current block is resident.
        tdm_load_2d_f16(k_off[cur ^ 1], Kbase + (ib + 1) * 64 * DHEAD,
                        DHEAD, 64, DHEAD, DHEAD, 64);
        tdm_load_2d_f16(v_off[cur ^ 1], Vbase + (ib + 1) * 64,
                        64, DHEAD, L_SEQ, 64, DHEAD);
        __builtin_amdgcn_s_wait_tensorcnt(2);
      } else {
        __builtin_amdgcn_s_wait_tensorcnt(0);
      }
    }
    __syncthreads();

    const half_t* kt = lds_k[cur];
    const half_t* vt = lds_v[cur];

#pragma unroll
    for (int kk = 0; kk < 64; kk += 32) {
      // ---- scores (log2 domain): two 16x16 tiles over 32 context columns ---
      v8f s0 = {};
      v8f s1 = {};
      s0 = wmma_f16(qa0, load_b_frag(kt + (kk + 0) * DHEAD + 0, DHEAD, lane), s0);
      s0 = wmma_f16(qa1, load_b_frag(kt + (kk + 0) * DHEAD + 32, DHEAD, lane), s0);
      s1 = wmma_f16(qa0, load_b_frag(kt + (kk + 16) * DHEAD + 0, DHEAD, lane), s1);
      s1 = wmma_f16(qa1, load_b_frag(kt + (kk + 16) * DHEAD + 32, DHEAD, lane), s1);

      // ---- online softmax via exp2 (row = VGPR index, 16-lane half-wave) ---
#pragma unroll
      for (int r = 0; r < 8; ++r) {
        float t = fmaxf(s0[r], s1[r]);
        t = fmaxf(t, __shfl_xor(t, 1));
        t = fmaxf(t, __shfl_xor(t, 2));
        t = fmaxf(t, __shfl_xor(t, 4));
        t = fmaxf(t, __shfl_xor(t, 8));
        float mnew = fmaxf(mr[r], t);
        float corr = exp2f(mr[r] - mnew);
        mr[r] = mnew;
        float p0 = exp2f(s0[r] - mnew);
        float p1 = exp2f(s1[r] - mnew);
        float rs = p0 + p1;
        rs += __shfl_xor(rs, 1);
        rs += __shfl_xor(rs, 2);
        rs += __shfl_xor(rs, 4);
        rs += __shfl_xor(rs, 8);
        lr[r] = lr[r] * corr + rs;
        o[0][r] *= corr; o[1][r] *= corr; o[2][r] *= corr; o[3][r] *= corr;
        s0[r] = p0; s1[r] = p1;
      }

      // ---- C-layout -> A-layout via per-wave LDS bounce --------------------
#pragma unroll
      for (int r = 0; r < 8; ++r) {
        pp[(mb + r) * 32 + n] = (half_t)s0[r];
        pp[(mb + r) * 32 + 16 + n] = (half_t)s1[r];
      }
      v16h pa = load_a_frag(pp, 32, lane);

      // ---- O += P @ V ------------------------------------------------------
#pragma unroll
      for (int nd = 0; nd < 4; ++nd) {
        o[nd] = wmma_f16(pa, load_b_frag(vt + (nd * 16) * 64 + kk, 64, lane),
                         o[nd]);
      }
    }
    __syncthreads();
  }

  // epilogue: normalize, write head-interleaved f16 [L][DMODEL]
  float inv[8];
#pragma unroll
  for (int r = 0; r < 8; ++r) inv[r] = 1.0f / lr[r];
#pragma unroll
  for (int nd = 0; nd < 4; ++nd) {
#pragma unroll
    for (int r = 0; r < 8; ++r) {
      int row = qbase + mb + r;
      int col = h * DHEAD + nd * 16 + n;
      Oh[row * DMODEL + col] = (half_t)(o[nd][r] * inv[r]);
    }
  }
}

// ---------------------------------------------------------------------------
// Kernel 3: out[4096,512] (f32) = Oh[4096,512] @ Wo  (WoT n-major f16)
// ---------------------------------------------------------------------------
__global__ __launch_bounds__(256) void proj_gemm_kernel(
    const half_t* __restrict__ Oh, const half_t* __restrict__ WoT,
    float* __restrict__ out) {
  const int NCH = DMODEL / 64;  // 8
  int wid = blockIdx.x * 8 + (threadIdx.x >> 5);
  int lane = threadIdx.x & 31;
  int mt = wid / NCH;
  int nc = wid % NCH;
  int row0 = mt * 16;

  v8f acc[4] = {{}, {}, {}, {}};
  for (int k0 = 0; k0 < DMODEL; k0 += 32) {
    v16h a = load_a_frag(Oh + row0 * DMODEL + k0, DMODEL, lane);
#pragma unroll
    for (int t = 0; t < 4; ++t) {
      v16h b = load_b_frag(WoT + (nc * 64 + t * 16) * DMODEL + k0, DMODEL, lane);
      acc[t] = wmma_f16(a, b, acc[t]);
    }
  }

  int n = lane & 15;
  int mb = (lane & 16) ? 8 : 0;
#pragma unroll
  for (int t = 0; t < 4; ++t) {
#pragma unroll
    for (int r = 0; r < 8; ++r) {
      out[(row0 + mb + r) * DMODEL + nc * 64 + t * 16 + n] = acc[t][r];
    }
  }
}

// ---------------------------------------------------------------------------
// Host launcher
// ---------------------------------------------------------------------------
extern "C" void kernel_launch(void* const* d_in, const int* in_sizes, int n_in,
                              void* d_out, int out_size, void* d_ws, size_t ws_size,
                              hipStream_t stream) {
  const float* x    = (const float*)d_in[0];   // [1,4096,512]
  const float* Wqkv = (const float*)d_in[1];   // [512,1536]
  const float* Wo   = (const float*)d_in[2];   // [512,512]
  float* out = (float*)d_out;

  half_t* ws  = (half_t*)d_ws;
  half_t* xh  = ws;                                  // 4096*512
  half_t* Wt  = xh  + L_SEQ * DMODEL;                // 1536*512
  half_t* WoT = Wt  + 3 * DMODEL * DMODEL;           // 512*512
  half_t* Qh  = WoT + DMODEL * DMODEL;               // 8*4096*64
  half_t* Kh  = Qh  + NHEAD * L_SEQ * DHEAD;
  half_t* VTh = Kh  + NHEAD * L_SEQ * DHEAD;
  half_t* Oh  = VTh + NHEAD * L_SEQ * DHEAD;         // 4096*512

  int nx = L_SEQ * DMODEL;
  cvt_f16_kernel<<<(nx + 255) / 256, 256, 0, stream>>>(x, xh, nx);
  cvt_tr_kernel<<<(DMODEL * 3 * DMODEL + 255) / 256, 256, 0, stream>>>(
      Wqkv, Wt, DMODEL, 3 * DMODEL);
  cvt_tr_kernel<<<(DMODEL * DMODEL + 255) / 256, 256, 0, stream>>>(
      Wo, WoT, DMODEL, DMODEL);

  // QKV: (4096/16) M-tiles * 24 N-chunks = 6144 waves / 8 waves-per-block
  qkv_gemm_kernel<<<768, 256, 0, stream>>>(xh, Wt, Qh, Kh, VTh);

  // Flash attention: grid (32 q-blocks, 8 heads)
  flash_attn_kernel<<<dim3(L_SEQ / 128, NHEAD), 256, 0, stream>>>(Qh, Kh, VTh, Oh);

  // Output projection: 256 M-tiles * 8 N-chunks = 2048 waves / 8 per block
  proj_gemm_kernel<<<256, 256, 0, stream>>>(Oh, WoT, out);
}
